// Agent_16595753632360
// MI455X (gfx1250) — matrix operand order repaired and checked by reading, not compiled, not run on hardware
//
#include <hip/hip_runtime.h>
#include <math.h>

// ---------------------------------------------------------------------------
// IMPALA agent forward on MI455X (gfx1250, wave32).
//  - Nature-CNN + FC as implicit-GEMM on v_wmma_f32_16x16x32_f16:
//      * A tiles double-buffered in LDS (gathers for chunk k+1 issued before
//        chunk k's WMMAs; one barrier per chunk) -> HBM latency hidden.
//      * A fragments = 2x ds_load_b128; B fragments = 2x global_load_b128
//        straight from L2-resident (N,K) f16 weights, WGP-scope prefetch.
//  - LSTM: sequential recurrence -> one persistent workgroup, WMMA gate GEMMs.
// Roofline: dominant HBM term is the 462 MB f32 input (~20 us @ 23.3 TB/s);
// all weights (~4 MB f16) and intermediates fit in the 192 MB L2.
// ---------------------------------------------------------------------------

typedef __attribute__((ext_vector_type(16))) _Float16 v16h;
typedef __attribute__((ext_vector_type(8)))  _Float16 v8h;
typedef __attribute__((ext_vector_type(8)))  float    v8f;

#define T_STEPS 128
#define B_ENVS  32
#define TBATCH  4096      // T*B
#define HID     128
#define FEAT    512
#define NACT    18

#define BM 128
#define BK 32
#define ASTR 48           // padded f16 stride for A tile rows (96 B, 16B aligned)

__device__ __forceinline__ float sigf(float x) { return 1.0f / (1.0f + expf(-x)); }

__device__ __forceinline__ v16h cat8(v8h a, v8h b) {
  return __builtin_shufflevector(a, b, 0, 1, 2, 3, 4, 5, 6, 7,
                                       8, 9, 10, 11, 12, 13, 14, 15);
}

// --------------------------- weight f32 -> f16 cast ------------------------
// Weights stay in their native (N,K) row-major layout: that is exactly what
// the WMMA B fragment wants (two runs of 8 consecutive K per lane-column -> b128 loads).
__global__ void wcast_kernel(const float* __restrict__ src,
                             _Float16* __restrict__ dst, int n) {
  int i = blockIdx.x * blockDim.x + threadIdx.x;
  if (i < n) dst[i] = (_Float16)src[i];
}

// ------------------- implicit-GEMM conv / FC with WMMA ---------------------
// out[M,N] = relu(im2col(src)[M,K] @ Wgt[N,K]^T + bias[N]), stored NCHW f16.
// Geometry is compile-time: no runtime division, no bounds guards
// (M % BM == 0 and N % BNT == 0 for every layer).
template <int C, int H, int W, int KH, int KW, int STR, int OH, int OW,
          int N, int K, int BNT, bool SRCF32, bool RELU>
__global__ __launch_bounds__(256)
void conv_gemm(const void* __restrict__ srcv, float scale,
               const _Float16* __restrict__ Wgt,   // (N,K) f16
               const float* __restrict__ bias,     // (N)
               _Float16* __restrict__ dst)         // NCHW f16
{
  constexpr int NSUB = BNT / 16;
  constexpr int KHKW = KH * KW;
  __shared__ alignas(16) _Float16 As[2][BM][ASTR];   // double-buffered A tile

  const int tid    = threadIdx.x;
  const int lane   = tid & 31;
  const int wave   = tid >> 5;
  const int blockM = blockIdx.x;
  const int blockN = blockIdx.y;

  const float*    srcF32 = (const float*)srcv;
  const _Float16* srcF16 = (const _Float16*)srcv;

  // A staging: this thread owns row (tid/2), halves (tid&1)*16 .. +15.
  const int arow  = tid >> 1;
  const int acol0 = (tid & 1) * 16;
  const int m_g   = blockM * BM + arow;
  const int n_img = m_g / (OH * OW);
  const int prem  = m_g - n_img * (OH * OW);
  const int oy    = prem / OW;
  const int ox    = prem - oy * OW;

  // im2col gather of 16 halves for K-chunk base kc (compile-time strides).
  auto gather16 = [&](int kc, _Float16* regs) {
#pragma unroll
    for (int i = 0; i < 16; ++i) {
      const int k  = kc + acol0 + i;
      const int ci = k / KHKW;
      const int q  = k - ci * KHKW;
      const int ky = q / KW;
      const int kx = q - ky * KW;
      const long idx = (((long)n_img * C + ci) * H + (oy * STR + ky)) * W +
                       (ox * STR + kx);
      regs[i] = SRCF32 ? (_Float16)(srcF32[idx] * scale) : srcF16[idx];
    }
  };

  // Fragment geometry (ISA 7.12.2, 16-bit A 16x32 / B 32x16 layouts).
  const int klo    = (lane < 16) ? 0 : 8;
  const int khi    = (lane < 16) ? 16 : 24;
  const int koff   = (lane < 16) ? 0 : 16;
  const int ncol_l = lane & 15;

  v8f acc[NSUB];
#pragma unroll
  for (int j = 0; j < NSUB; ++j)
    acc[j] = (v8f){0.f, 0.f, 0.f, 0.f, 0.f, 0.f, 0.f, 0.f};

  // prologue: stage chunk 0 into buffer 0
  {
    _Float16 regs[16];
    gather16(0, regs);
#pragma unroll
    for (int i = 0; i < 16; ++i) As[0][arow][acol0 + i] = regs[i];
  }
  __syncthreads();

  int buf = 0;
  for (int k0 = 0; k0 < K; k0 += BK) {
    const bool have_next = (k0 + BK) < K;   // uniform

    // (1) issue next chunk's gathers early (overlaps this chunk's WMMAs)
    _Float16 regs[16];
    if (have_next) gather16(k0 + BK, regs);

    // (2) compute from As[buf]
    const _Float16* arowp = &As[buf][wave * 16 + (lane & 15)][0];
    const v8h a0 = *(const v8h*)(arowp + klo);
    const v8h a1 = *(const v8h*)(arowp + khi);
    const v16h af = cat8(a0, a1);
#pragma unroll
    for (int j = 0; j < NSUB; ++j) {
      const int n_g = blockN * BNT + j * 16 + ncol_l;
      const _Float16* bp = Wgt + (size_t)n_g * K + (k0 + koff);
      const v8h b0 = *(const v8h*)(Wgt + (size_t)n_g * K + k0 + klo);
      const v8h b1 = *(const v8h*)(Wgt + (size_t)n_g * K + k0 + khi);
      if (have_next) __builtin_prefetch(bp + BK, 0, 3);  // WGP-scope prefetch
      const v16h bf = cat8(b0, b1);
      acc[j] = __builtin_amdgcn_wmma_f32_16x16x32_f16(
          false, af, false, bf, (short)0, acc[j], false, false);
    }

    // (3) commit staged registers to the alternate buffer
    if (have_next) {
#pragma unroll
      for (int i = 0; i < 16; ++i) As[buf ^ 1][arow][acol0 + i] = regs[i];
    }
    __syncthreads();
    buf ^= 1;
  }

  // ---- epilogue: bias + relu, scatter to NCHW f16 ----
  const int row_l = (lane < 16) ? 0 : 8;
#pragma unroll
  for (int j = 0; j < NSUB; ++j) {
    const int n_g = blockN * BNT + j * 16 + ncol_l;
    const float bv = bias[n_g];
#pragma unroll
    for (int r = 0; r < 8; ++r) {
      const int m_row = blockM * BM + wave * 16 + row_l + r;
      float v = acc[j][r] + bv;
      if (RELU) v = fmaxf(v, 0.0f);
      const int ni  = m_row / (OH * OW);
      const int pp  = m_row - ni * (OH * OW);
      const int oyy = pp / OW;
      const int oxx = pp - oyy * OW;
      const long didx = (((long)ni * N + n_g) * OH + oyy) * OW + oxx;
      dst[didx] = (_Float16)v;
    }
  }
}

// ------------------------ persistent LSTM kernel ---------------------------
// One workgroup (8 waves). Per step: gates = X_t @ Wih^T + h @ Whh^T + b as
// 64 WMMA 16x16 tiles; W kept in (N,K) f16 layout -> contiguous B fragments.
__global__ __launch_bounds__(256)
void lstm_kernel(const _Float16* __restrict__ feat,   // (TB, FEAT) f16
                 const int* __restrict__ done,        // (TB)
                 const float* __restrict__ h0,        // (B,HID)
                 const float* __restrict__ c0,        // (B,HID)
                 const _Float16* __restrict__ Wih,    // (4*HID, FEAT) f16
                 const _Float16* __restrict__ Whh,    // (4*HID, HID) f16
                 const float* __restrict__ b_ih,      // (4*HID)
                 const float* __restrict__ b_hh,      // (4*HID)
                 float* __restrict__ hidden)          // (TB, HID) f32
{
  extern __shared__ char smem[];
  _Float16(*Xs)[FEAT] = (_Float16(*)[FEAT])smem;                          // 32 KB
  _Float16(*Hs)[HID]  = (_Float16(*)[HID])(smem + 32768);                 // 8 KB
  float(*Cs)[HID]     = (float(*)[HID])(smem + 32768 + 8192);             // 16 KB
  float(*Gs)[4 * HID] = (float(*)[4 * HID])(smem + 32768 + 8192 + 16384); // 64 KB

  const int tid  = threadIdx.x;
  const int lane = tid & 31;
  const int wave = tid >> 5;

  const int klo    = (lane < 16) ? 0 : 8;
  const int khi    = (lane < 16) ? 16 : 24;
  const int koff   = (lane < 16) ? 0 : 16;
  const int ncol_l = lane & 15;

  for (int i = tid; i < B_ENVS * HID; i += 256) {
    int b = i / HID, u = i - b * HID;
    Hs[b][u] = (_Float16)h0[i];
    Cs[b][u] = c0[i];
  }
  __syncthreads();

  for (int t = 0; t < T_STEPS; ++t) {
    // done-mask reset of recurrent state
    for (int i = tid; i < B_ENVS * HID; i += 256) {
      int b = i / HID, u = i - b * HID;
      float m = 1.0f - (float)done[t * B_ENVS + b];
      Hs[b][u] = (_Float16)((float)Hs[b][u] * m);
      Cs[b][u] *= m;
    }
    // stage X_t tile
    for (int i = tid; i < B_ENVS * FEAT; i += 256) {
      int b = i / FEAT, u = i - b * FEAT;
      Xs[b][u] = feat[(long)(t * B_ENVS + b) * FEAT + u];
    }
    __syncthreads();

    // 64 gate tiles (2 M-tiles x 32 N-tiles), 8 per wave
    for (int tt = wave; tt < 64; tt += 8) {
      const int mt   = tt >> 5;
      const int nt   = tt & 31;
      const int row  = mt * 16 + (lane & 15);
      const int ncol = nt * 16 + ncol_l;
      v8f acc = (v8f){0.f, 0.f, 0.f, 0.f, 0.f, 0.f, 0.f, 0.f};

      const _Float16* xrow = &Xs[row][0];
      const _Float16* brow = Wih + (size_t)ncol * FEAT;
      for (int k0 = 0; k0 < FEAT; k0 += BK) {
        const v16h af = cat8(*(const v8h*)(xrow + k0 + klo),
                             *(const v8h*)(xrow + k0 + khi));
        const _Float16* bp = brow + k0 + koff;
        if (k0 + BK < FEAT) __builtin_prefetch(bp + BK, 0, 3);  // WGP scope
        const v16h bf = cat8(*(const v8h*)(brow + k0 + klo), *(const v8h*)(brow + k0 + khi));
        acc = __builtin_amdgcn_wmma_f32_16x16x32_f16(
            false, af, false, bf, (short)0, acc, false, false);
      }
      const _Float16* hrow = &Hs[row][0];
      const _Float16* crow = Whh + (size_t)ncol * HID;
      for (int k0 = 0; k0 < HID; k0 += BK) {
        const v16h af = cat8(*(const v8h*)(hrow + k0 + klo),
                             *(const v8h*)(hrow + k0 + khi));
        const _Float16* bp = crow + k0 + koff;
        const v16h bf = cat8(*(const v8h*)(crow + k0 + klo), *(const v8h*)(crow + k0 + khi));
        acc = __builtin_amdgcn_wmma_f32_16x16x32_f16(
            false, af, false, bf, (short)0, acc, false, false);
      }
      const float bsum  = b_ih[ncol] + b_hh[ncol];
      const int   rbase = mt * 16 + ((lane < 16) ? 0 : 8);
#pragma unroll
      for (int r = 0; r < 8; ++r) Gs[rbase + r][ncol] = acc[r] + bsum;
    }
    __syncthreads();

    // elementwise LSTM cell (torch gate order i,f,g,o)
    for (int i = tid; i < B_ENVS * HID; i += 256) {
      int b = i / HID, u = i - b * HID;
      float ig = Gs[b][u];
      float fg = Gs[b][HID + u];
      float gg = Gs[b][2 * HID + u];
      float og = Gs[b][3 * HID + u];
      float c  = sigf(fg) * Cs[b][u] + sigf(ig) * tanhf(gg);
      float h  = sigf(og) * tanhf(c);
      Cs[b][u] = c;
      Hs[b][u] = (_Float16)h;
      hidden[(long)(t * B_ENVS + b) * HID + u] = h;
    }
    __syncthreads();
  }
}

// ------------------------------- heads -------------------------------------
__global__ void heads_kernel(const float* __restrict__ hidden,
                             const float* __restrict__ wa, const float* __restrict__ ba,
                             const float* __restrict__ wc, const float* __restrict__ bc,
                             float* __restrict__ out) {
  int idx = blockIdx.x * blockDim.x + threadIdx.x;
  const int total = TBATCH * (NACT + 1);
  if (idx >= total) return;
  int m = idx / (NACT + 1);
  int j = idx - m * (NACT + 1);
  const float* w = (j < NACT) ? (wa + (long)j * HID) : wc;
  float bias = (j < NACT) ? ba[j] : bc[0];
  float s = 0.0f;
#pragma unroll 4
  for (int u = 0; u < HID; ++u) s += hidden[(long)m * HID + u] * w[u];
  out[idx] = s + bias;
}

// ------------------------------ launcher ------------------------------------
extern "C" void kernel_launch(void* const* d_in, const int* in_sizes, int n_in,
                              void* d_out, int out_size, void* d_ws, size_t ws_size,
                              hipStream_t stream) {
  (void)in_sizes; (void)n_in; (void)out_size; (void)ws_size;
  const float* x    = (const float*)d_in[0];
  const int*   done = (const int*)d_in[1];
  const float* h0   = (const float*)d_in[2];
  const float* c0   = (const float*)d_in[3];
  const float* w1   = (const float*)d_in[4];
  const float* b1   = (const float*)d_in[5];
  const float* w2   = (const float*)d_in[6];
  const float* b2   = (const float*)d_in[7];
  const float* w3   = (const float*)d_in[8];
  const float* b3   = (const float*)d_in[9];
  const float* wf   = (const float*)d_in[10];
  const float* bf   = (const float*)d_in[11];
  const float* w_ih = (const float*)d_in[12];
  const float* w_hh = (const float*)d_in[13];
  const float* b_ih = (const float*)d_in[14];
  const float* b_hh = (const float*)d_in[15];
  const float* wa   = (const float*)d_in[16];
  const float* ba   = (const float*)d_in[17];
  const float* wc   = (const float*)d_in[18];
  const float* bc   = (const float*)d_in[19];
  float* out = (float*)d_out;

  // workspace carving (256B aligned)
  char* p = (char*)d_ws;
  auto carve = [&](size_t bytes) -> char* {
    char* r = p;
    p += (bytes + 255) & ~(size_t)255;
    return r;
  };
  _Float16* c1   = (_Float16*)carve((size_t)TBATCH * 32 * 20 * 20 * 2);
  _Float16* c2   = (_Float16*)carve((size_t)TBATCH * 64 * 9 * 9 * 2);
  _Float16* c3   = (_Float16*)carve((size_t)TBATCH * 64 * 7 * 7 * 2);
  _Float16* feat = (_Float16*)carve((size_t)TBATCH * FEAT * 2);
  float*    hid  = (float*)carve((size_t)TBATCH * HID * 4);
  _Float16* W1   = (_Float16*)carve((size_t)32 * 256 * 2);
  _Float16* W2   = (_Float16*)carve((size_t)64 * 512 * 2);
  _Float16* W3   = (_Float16*)carve((size_t)64 * 576 * 2);
  _Float16* Wf   = (_Float16*)carve((size_t)512 * 3136 * 2);
  _Float16* Wih  = (_Float16*)carve((size_t)512 * 512 * 2);
  _Float16* Whh  = (_Float16*)carve((size_t)512 * 128 * 2);

  auto cast_launch = [&](const float* Wm, _Float16* Bm, int n) {
    wcast_kernel<<<dim3((n + 255) / 256), dim3(256), 0, stream>>>(Wm, Bm, n);
  };
  cast_launch(w1, W1, 32 * 256);
  cast_launch(w2, W2, 64 * 512);
  cast_launch(w3, W3, 64 * 576);
  cast_launch(wf, Wf, 512 * 3136);
  cast_launch(w_ih, Wih, 512 * 512);
  cast_launch(w_hh, Whh, 512 * 128);

  // conv1: (TB,4,84,84) -> (TB,32,20,20)  M=1638400 N=32 K=256
  conv_gemm<4, 84, 84, 8, 8, 4, 20, 20, 32, 256, 32, true, true>
      <<<dim3(1638400 / BM, 1), dim3(256), 0, stream>>>(
          (const void*)x, 1.0f / 255.0f, W1, b1, c1);
  // conv2: -> (TB,64,9,9)  M=331776 N=64 K=512
  conv_gemm<32, 20, 20, 4, 4, 2, 9, 9, 64, 512, 64, false, true>
      <<<dim3(331776 / BM, 1), dim3(256), 0, stream>>>(
          (const void*)c1, 1.0f, W2, b2, c2);
  // conv3: -> (TB,64,7,7)  M=200704 N=64 K=576
  conv_gemm<64, 9, 9, 3, 3, 1, 7, 7, 64, 576, 64, false, true>
      <<<dim3(200704 / BM, 1), dim3(256), 0, stream>>>(
          (const void*)c2, 1.0f, W3, b3, c3);
  // FC: (TB,3136) @ (3136,512) as 1x1 "conv"  M=4096 N=512 K=3136
  conv_gemm<3136, 1, 1, 1, 1, 1, 1, 1, 512, 3136, 64, false, true>
      <<<dim3(TBATCH / BM, 512 / 64), dim3(256), 0, stream>>>(
          (const void*)c3, 1.0f, Wf, bf, feat);

  // LSTM: single persistent workgroup, 120 KB dynamic LDS
  size_t lds = 32768 + 8192 + 16384 + 65536;
  lstm_kernel<<<dim3(1), dim3(256), lds, stream>>>(
      feat, done, h0, c0, Wih, Whh, b_ih, b_hh, hid);

  // heads: (TB, 19)
  heads_kernel<<<dim3((TBATCH * (NACT + 1) + 255) / 256), dim3(256), 0, stream>>>(
      hid, wa, ba, wc, bc, out);
}
